// Attention_27925877358889
// MI455X (gfx1250) — matrix operand order repaired and checked by reading, not compiled, not run on hardware
//
#include <hip/hip_runtime.h>
#include <hip/hip_bf16.h>

// ---------------------------------------------------------------------------
// Problem constants (match reference)
// ---------------------------------------------------------------------------
#define HIDDEN   1024
#define HEADS    16
#define HEAD_DIM 64
#define SEQ      2048
#define BATCH    4
#define TOKENS   (BATCH * SEQ)          // 8192

typedef __attribute__((ext_vector_type(16))) __bf16 bf16x16;   // WMMA A/B operand
typedef __attribute__((ext_vector_type(8)))  float  f32x8;     // WMMA C/D operand

union ABFrag  { uint4 u[2]; bf16x16 v; };   // 32 bytes
union AccFrag { f32x8 v; float f[8]; };

// CDNA5 async global->LDS path (builtin params are v4i32 in AS1 / AS3).
#if defined(__has_builtin)
#  if __has_builtin(__builtin_amdgcn_global_load_async_to_lds_b128)
#    define USE_ASYNC_LDS 1
#  endif
#endif
#ifndef USE_ASYNC_LDS
#  define USE_ASYNC_LDS 0
#endif

typedef int v4i __attribute__((vector_size(16)));
typedef __attribute__((address_space(1))) v4i as1_v4i;   // global
typedef __attribute__((address_space(3))) v4i as3_v4i;   // LDS

__device__ __forceinline__ void async_copy16(const __hip_bfloat16* g,
                                             __hip_bfloat16* l) {
#if USE_ASYNC_LDS
    __builtin_amdgcn_global_load_async_to_lds_b128((as1_v4i*)g, (as3_v4i*)l, 0, 0);
#else
    *reinterpret_cast<uint4*>(l) = *reinterpret_cast<const uint4*>(g);
#endif
}

__device__ __forceinline__ void wait_asynccnt0() {
#if USE_ASYNC_LDS
#  if defined(__has_builtin)
#    if __has_builtin(__builtin_amdgcn_s_wait_asynccnt)
    __builtin_amdgcn_s_wait_asynccnt(0);
#    else
    asm volatile("s_wait_asynccnt 0" ::: "memory");
#    endif
#  else
    asm volatile("s_wait_asynccnt 0" ::: "memory");
#  endif
#endif
}

// Load a 32-byte bf16 fragment as two contiguous 16-byte chunks.
__device__ __forceinline__ bf16x16 load_frag2(const __hip_bfloat16* p0,
                                              const __hip_bfloat16* p1) {
    ABFrag r;
    r.u[0] = *reinterpret_cast<const uint4*>(p0);
    r.u[1] = *reinterpret_cast<const uint4*>(p1);
    return r.v;
}

// ---------------------------------------------------------------------------
// fp32 -> bf16 conversion (grid-stride)
// ---------------------------------------------------------------------------
__global__ void cvt_f32_bf16(const float* __restrict__ in,
                             __hip_bfloat16* __restrict__ out, int n) {
    int i = blockIdx.x * blockDim.x + threadIdx.x;
    int stride = gridDim.x * blockDim.x;
    for (; i < n; i += stride) out[i] = __float2bfloat16(in[i]);
}

// ---------------------------------------------------------------------------
// Templated WMMA GEMM:  y[m,e] = sum_d A[m,d] * W[e,d] + bias[e]
// W tiles (64 e x 64 k) double-buffered in LDS via async DMA, shared by the
// block's 4 waves; async issue precedes compute so the DMA latency hides
// behind 8 WMMAs per stage.
//   MODE 0: V path  -> write V^T bf16 [B,H,D,S]
//   MODE 1: K path  -> in-register RoPE, write bf16 [B,H,S,D]
//   MODE 2: Q path  -> RoPE + 1/sqrt(D) scale, write bf16 [B,H,S,D]
//   MODE 3: output  -> write fp32 [M x 1024]
// ---------------------------------------------------------------------------
template <int MODE>
__global__ __launch_bounds__(128) void gemm_rope(
    const __hip_bfloat16* __restrict__ A,
    const __hip_bfloat16* __restrict__ W,
    const float* __restrict__ bias,
    __hip_bfloat16* __restrict__ outb,
    float* __restrict__ outf) {

    __shared__ __align__(16) __hip_bfloat16 wsm[2][64][72];   // 18 KB, padded rows

    const int tid   = threadIdx.x;
    const int lane  = tid & 31;
    const int wave  = tid >> 5;
    const int half  = lane >> 4;
    const int n15   = lane & 15;
    const int mBase = (blockIdx.x * 4 + wave) * 16;
    const int nBase = blockIdx.y * 64;

    const __hip_bfloat16* aRow = A + (size_t)(mBase + n15) * HIDDEN;

    // stage a 64x64 W tile (rows e = nBase.., cols k0..k0+63): 4 b128/thread
    auto stageW = [&](int k0, int bufi) {
#pragma unroll
        for (int it = 0; it < 4; ++it) {
            int idx = it * 128 + tid;
            int row = idx >> 3, ch = idx & 7;
            async_copy16(W + (size_t)(nBase + row) * HIDDEN + k0 + ch * 8,
                         &wsm[bufi][row][ch * 8]);
        }
    };

    AccFrag acc[4];
#pragma unroll
    for (int f = 0; f < 4; ++f)
#pragma unroll
        for (int r = 0; r < 8; ++r) acc[f].f[r] = 0.0f;

    int buf = 0;
    stageW(0, 0);
    for (int k0 = 0; k0 < HIDDEN; k0 += 64) {
        wait_asynccnt0();          // my portion of buf has landed
        __syncthreads();           // everyone's portion has landed
        if (k0 + 64 < HIDDEN) stageW(k0 + 64, buf ^ 1);  // prefetch next stage
#pragma unroll
        for (int kh = 0; kh < 2; ++kh) {
            bf16x16 af = load_frag2(aRow + k0 + kh * 32 + 16 * half,
                                    aRow + k0 + kh * 32 + 16 * half + 8);
#pragma unroll
            for (int f = 0; f < 4; ++f) {
                bf16x16 bfv = load_frag2(&wsm[buf][f * 16 + n15][kh * 32 + 16 * half],
                                         &wsm[buf][f * 16 + n15][kh * 32 + 16 * half + 8]);
                acc[f].v = __builtin_amdgcn_wmma_f32_16x16x32_bf16(
                    false, af, false, bfv, (short)0, acc[f].v, false, false);
            }
        }
        buf ^= 1;
    }

#pragma unroll
    for (int f = 0; f < 4; ++f) {
        float bv = bias[nBase + f * 16 + n15];
#pragma unroll
        for (int r = 0; r < 8; ++r) acc[f].f[r] += bv;
    }

    if (MODE == 3) {  // fp32 output, [M x HIDDEN]
#pragma unroll
        for (int f = 0; f < 4; ++f)
#pragma unroll
            for (int r = 0; r < 8; ++r) {
                int row = mBase + r + 8 * half;
                outf[(size_t)row * HIDDEN + nBase + f * 16 + n15] = acc[f].f[r];
            }
        return;
    }

    const int h = nBase >> 6;  // 64-wide strip == one head

    if (MODE == 0) {  // V: store transposed V^T[b,h,d,s]
#pragma unroll
        for (int f = 0; f < 4; ++f) {
            int d = f * 16 + n15;
#pragma unroll
            for (int r = 0; r < 8; ++r) {
                int tok = mBase + r + 8 * half;
                int b = tok >> 11, s = tok & (SEQ - 1);
                outb[((size_t)((b * HEADS + h) * HEAD_DIM + d)) * SEQ + s] =
                    __float2bfloat16(acc[f].f[r]);
            }
        }
        return;
    }

    // MODE 1/2: RoPE within the 64-wide head strip (in registers, via shfl)
    {
        const float LN_BASE = 9.210340371976184f;  // ln(10000)
        float nw[4][8];
#pragma unroll
        for (int r = 0; r < 8; ++r) {
            int tok = mBase + r + 8 * half;
            int s   = tok & (SEQ - 1);
            float cs[2], sn[2];
#pragma unroll
            for (int jj = 0; jj < 2; ++jj) {
                int j = jj * 16 + n15;
                float inv = __expf(-(2.0f * (float)j / (float)HEAD_DIM) * LN_BASE);
                __sincosf((float)s * inv, &sn[jj], &cs[jj]);
            }
#pragma unroll
            for (int f = 0; f < 4; ++f) {
                int jj = f & 1;
                int j  = jj * 16 + n15;
                int a  = 2 * j;
                int fb = jj * 2;
                int sel = (a >> 4) & 1;
                int la  = (a & 15) | (lane & 16);
                float x1l = __shfl(acc[fb].f[r],     la,     32);
                float x1h = __shfl(acc[fb + 1].f[r], la,     32);
                float x2l = __shfl(acc[fb].f[r],     la + 1, 32);
                float x2h = __shfl(acc[fb + 1].f[r], la + 1, 32);
                float x1 = sel ? x1h : x1l;
                float x2 = sel ? x2h : x2l;
                float o = (f < 2) ? (x1 * cs[jj] - x2 * sn[jj])
                                  : (x1 * sn[jj] + x2 * cs[jj]);
                if (MODE == 2) o *= 0.125f;
                nw[f][r] = o;
            }
        }
#pragma unroll
        for (int f = 0; f < 4; ++f)
#pragma unroll
            for (int r = 0; r < 8; ++r) {
                int tok = mBase + r + 8 * half;
                int b = tok >> 11, s = tok & (SEQ - 1);
                int d = f * 16 + n15;
                outb[((size_t)((b * HEADS + h) * SEQ + s)) * HEAD_DIM + d] =
                    __float2bfloat16(nw[f][r]);
            }
    }
}

// ---------------------------------------------------------------------------
// Flash attention, fully transposed inner math:
//   S^T = K . Q^T   (rows = keys t, cols = queries s)
//   O^T = V^T . P^T (rows = d, cols = queries s)
// Per-query softmax stats are per-lane scalars (in-lane trees + one xor16).
// K / V^T tiles are double-buffered in LDS via async DMA: chunk i+1's loads
// are issued before chunk i's compute, so DMA latency hides behind WMMA work.
// One block barrier per chunk.
// ---------------------------------------------------------------------------
__global__ __launch_bounds__(256) void attn_fused(
    const __hip_bfloat16* __restrict__ Q,   // [B,H,S,D]
    const __hip_bfloat16* __restrict__ K,   // [B,H,S,D]
    const __hip_bfloat16* __restrict__ VT,  // [B,H,D,S]
    __hip_bfloat16* __restrict__ O) {       // [B,S,HIDDEN]

    // per-wave P^T / O^T scratch: 16 query rows x 72 (padded) cols
    __shared__ __align__(16) __hip_bfloat16 plds[8][16][72];      // 18 KB
    // double-buffered staged tiles (padded strides: 144B / 80B rows)
    __shared__ __align__(16) __hip_bfloat16 ks[2][32][72];        // K:  32 t x 64 d
    __shared__ __align__(16) __hip_bfloat16 vs[2][64][40];        // VT: 64 d x 32 t

    const int tid  = threadIdx.x;
    const int lane = tid & 31;
    const int wave = tid >> 5;
    const int half = lane >> 4;
    const int n15  = lane & 15;
    const int bh   = blockIdx.x >> 4;       // b*HEADS + h
    const int sc   = blockIdx.x & 15;
    const int sTile = sc * 128 + wave * 16;

    const __hip_bfloat16* qb = Q  + (size_t)bh * SEQ * HEAD_DIM;
    const __hip_bfloat16* kb = K  + (size_t)bh * SEQ * HEAD_DIM;
    const __hip_bfloat16* vb = VT + (size_t)bh * HEAD_DIM * SEQ;

    // staging indices (one b128 per thread per tile)
    const int krow = tid >> 3, kch = tid & 7;   // K: 32 rows x 8 chunks
    const int vrow = tid >> 2, vch = tid & 3;   // VT: 64 rows x 4 chunks
    auto stageKV = [&](int t0, int bufi) {
        async_copy16(kb + (size_t)(t0 + krow) * HEAD_DIM + kch * 8,
                     &ks[bufi][krow][kch * 8]);
        async_copy16(vb + (size_t)vrow * SEQ + t0 + vch * 8,
                     &vs[bufi][vrow][vch * 8]);
    };

    // Preload Q as B-operand fragments (B[kk=d][n=s] = Q[s][d]), 2 K-steps
    bf16x16 qf[2];
#pragma unroll
    for (int s2 = 0; s2 < 2; ++s2) {
        const __hip_bfloat16* qr =
            qb + (size_t)(sTile + n15) * HEAD_DIM + s2 * 32 + 16 * half;
        qf[s2] = load_frag2(qr, qr + 8);
    }

    AccFrag o[4];     // O^T tiles: rows d = f*16 + r + 8*half, col s = n15
#pragma unroll
    for (int f = 0; f < 4; ++f)
#pragma unroll
        for (int r = 0; r < 8; ++r) o[f].f[r] = 0.0f;
    float mrow = -1e30f, lrow = 0.0f;   // per-lane: stats of query s = sTile+n15

    int buf = 0;
    stageKV(0, 0);
    for (int t0 = 0; t0 < SEQ; t0 += 32) {
        wait_asynccnt0();          // my chunk-t0 DMA done
        __syncthreads();           // everyone's done; previous buf free for reuse
        if (t0 + 32 < SEQ) stageKV(t0 + 32, buf ^ 1);   // prefetch next chunk

        // ---- S^T[32 x 16] = K(32x64) . Q^T : A = K rows (from LDS) ----
        AccFrag st[2];
#pragma unroll
        for (int tn = 0; tn < 2; ++tn) {
#pragma unroll
            for (int r = 0; r < 8; ++r) st[tn].f[r] = 0.0f;
#pragma unroll
            for (int s2 = 0; s2 < 2; ++s2) {
                bf16x16 kf = load_frag2(
                    &ks[buf][tn * 16 + n15][s2 * 32 + 16 * half],
                    &ks[buf][tn * 16 + n15][s2 * 32 + 16 * half + 8]);
                st[tn].v = __builtin_amdgcn_wmma_f32_16x16x32_bf16(
                    false, kf, false, qf[s2], (short)0, st[tn].v, false, false);
            }
        }

        // ---- online softmax: 16 in-lane values + one xor16 per reduction ----
        float mx = -1e30f;
#pragma unroll
        for (int tn = 0; tn < 2; ++tn)
#pragma unroll
            for (int r = 0; r < 8; ++r) mx = fmaxf(mx, st[tn].f[r]);
        mx = fmaxf(mx, __shfl_xor(mx, 16, 32));
        float mnew  = fmaxf(mrow, mx);
        float alpha = __expf(mrow - mnew);
        float rs = 0.0f;
#pragma unroll
        for (int tn = 0; tn < 2; ++tn)
#pragma unroll
            for (int r = 0; r < 8; ++r) {
                float p = __expf(st[tn].f[r] - mnew);
                rs += p;
                // P^T to LDS: row = query s (n15), col = key t
                plds[wave][n15][tn * 16 + r + 8 * half] = __float2bfloat16(p);
            }
        rs += __shfl_xor(rs, 16, 32);
        lrow = lrow * alpha + rs;
        mrow = mnew;
#pragma unroll
        for (int f = 0; f < 4; ++f)
#pragma unroll
            for (int r = 0; r < 8; ++r) o[f].f[r] *= alpha;

        // wave-private LDS region; fence compiler + DS counter
        asm volatile("s_wait_dscnt 0" ::: "memory");

        // ---- P^T as B-operand: lane n = s, kk = t (contiguous in LDS row) --
        bf16x16 pf = load_frag2(&plds[wave][n15][16 * half],
                                &plds[wave][n15][16 * half + 8]);

        // ---- O^T += V^T(64x32) . P^T(32x16): A = V^T rows (from LDS) ----
#pragma unroll
        for (int f = 0; f < 4; ++f) {
            bf16x16 vf = load_frag2(&vs[buf][f * 16 + n15][16 * half],
                                    &vs[buf][f * 16 + n15][16 * half + 8]);
            o[f].v = __builtin_amdgcn_wmma_f32_16x16x32_bf16(
                false, vf, false, pf, (short)0, o[f].v, false, false);
        }
        buf ^= 1;
    }

    // ---- normalize (per-lane), transpose via LDS, coalesced b128 stores ----
    const float inv = 1.0f / lrow;
#pragma unroll
    for (int f = 0; f < 4; ++f)
#pragma unroll
        for (int r = 0; r < 8; ++r)
            plds[wave][n15][f * 16 + r + 8 * half] =
                __float2bfloat16(o[f].f[r] * inv);
    asm volatile("s_wait_dscnt 0" ::: "memory");

    const int b = bh >> 4, h = bh & 15;
#pragma unroll
    for (int it = 0; it < 4; ++it) {
        int idx = it * 32 + lane;
        int row = idx >> 3;          // query row within tile
        int ch  = idx & 7;           // 8-element chunk of the 64-wide head
        uint4 t = *reinterpret_cast<const uint4*>(&plds[wave][row][ch * 8]);
        *reinterpret_cast<uint4*>(
            &O[(size_t)(b * SEQ + sTile + row) * HIDDEN + h * HEAD_DIM + ch * 8]) = t;
    }
}

// ---------------------------------------------------------------------------
// Launch: cvt -> QKV GEMMs (+RoPE) -> flash attention -> output GEMM
// ---------------------------------------------------------------------------
extern "C" void kernel_launch(void* const* d_in, const int* in_sizes, int n_in,
                              void* d_out, int out_size, void* d_ws, size_t ws_size,
                              hipStream_t stream) {
    const float* x  = (const float*)d_in[0];
    const float* Wq = (const float*)d_in[1];
    const float* bq = (const float*)d_in[2];
    const float* Wk = (const float*)d_in[3];
    const float* bk = (const float*)d_in[4];
    const float* Wv = (const float*)d_in[5];
    const float* bv = (const float*)d_in[6];
    const float* Wo = (const float*)d_in[7];
    const float* bo = (const float*)d_in[8];
    float* out = (float*)d_out;

    char* ws = (char*)d_ws;
    size_t off = 0;
    auto carve = [&](size_t bytes) {
        void* p = ws + off;
        off += (bytes + 255) & ~(size_t)255;
        return p;
    };
    const size_t NTOK_EL = (size_t)TOKENS * HIDDEN;          // 8,388,608
    __hip_bfloat16* xbf  = (__hip_bfloat16*)carve(NTOK_EL * 2);
    __hip_bfloat16* wqb  = (__hip_bfloat16*)carve((size_t)HIDDEN * HIDDEN * 2);
    __hip_bfloat16* wkb  = (__hip_bfloat16*)carve((size_t)HIDDEN * HIDDEN * 2);
    __hip_bfloat16* wvb  = (__hip_bfloat16*)carve((size_t)HIDDEN * HIDDEN * 2);
    __hip_bfloat16* wob  = (__hip_bfloat16*)carve((size_t)HIDDEN * HIDDEN * 2);
    __hip_bfloat16* qbuf = (__hip_bfloat16*)carve(NTOK_EL * 2);  // [B,H,S,D]
    __hip_bfloat16* kbuf = (__hip_bfloat16*)carve(NTOK_EL * 2);  // [B,H,S,D]
    __hip_bfloat16* vT   = (__hip_bfloat16*)carve(NTOK_EL * 2);  // [B,H,D,S]
    __hip_bfloat16* attn = (__hip_bfloat16*)carve(NTOK_EL * 2);  // [B,S,HIDDEN]

    cvt_f32_bf16<<<2048, 256, 0, stream>>>(x,  xbf, (int)NTOK_EL);
    cvt_f32_bf16<<<1024, 256, 0, stream>>>(Wq, wqb, HIDDEN * HIDDEN);
    cvt_f32_bf16<<<1024, 256, 0, stream>>>(Wk, wkb, HIDDEN * HIDDEN);
    cvt_f32_bf16<<<1024, 256, 0, stream>>>(Wv, wvb, HIDDEN * HIDDEN);
    cvt_f32_bf16<<<1024, 256, 0, stream>>>(Wo, wob, HIDDEN * HIDDEN);

    dim3 g(TOKENS / 64, HIDDEN / 64);  // 128 x 16 blocks, 128 threads (4 waves)
    gemm_rope<2><<<g, 128, 0, stream>>>(xbf, wqb, bq, qbuf, nullptr);  // Q: rope+scale
    gemm_rope<1><<<g, 128, 0, stream>>>(xbf, wkb, bk, kbuf, nullptr);  // K: rope
    gemm_rope<0><<<g, 128, 0, stream>>>(xbf, wvb, bv, vT,   nullptr);  // V -> V^T

    attn_fused<<<BATCH * HEADS * (SEQ / 128), 256, 0, stream>>>(qbuf, kbuf, vT, attn);

    gemm_rope<3><<<g, 128, 0, stream>>>(attn, wob, bo, nullptr, out);  // out proj fp32
}
